// Trend_39024072851793
// MI455X (gfx1250) — matrix operand order repaired and checked by reading, not compiled, not run on hardware
//
#include <hip/hip_runtime.h>

// ---------------------------------------------------------------------------
// Types / feature probes
// ---------------------------------------------------------------------------
typedef __bf16 bf16_t;
typedef __attribute__((ext_vector_type(16))) __bf16 v16bf;
typedef __attribute__((ext_vector_type(8)))  __bf16 v8bf;
typedef __attribute__((ext_vector_type(8)))  float  v8f;
typedef __attribute__((ext_vector_type(4)))  float  v4f;
typedef __attribute__((ext_vector_type(4)))  int    v4i;

// async-copy builtin takes int4 pointers: AS(1) source, AS(3) LDS dest
typedef __attribute__((address_space(1))) v4i as1_v4i;
typedef __attribute__((address_space(3))) v4i as3_v4i;

#if defined(__has_builtin)
# if __has_builtin(__builtin_amdgcn_global_load_async_to_lds_b128) && \
     __has_builtin(__builtin_amdgcn_s_wait_asynccnt)
#  define HAVE_ASYNC_LDS 1
# endif
#endif
#ifndef HAVE_ASYNC_LDS
# define HAVE_ASYNC_LDS 0
#endif

#define LDL 40   // LDS row pitch in bf16 elements (80B: 16B-aligned, bank-skewed)

// ---------------------------------------------------------------------------
// Transpose + cast fp32[R,C] -> bf16[C,R], batched over grid.z
// ---------------------------------------------------------------------------
__global__ __launch_bounds__(256)
void transpose_cast(const float* __restrict__ in, bf16_t* __restrict__ out,
                    int R, int C) {
  __shared__ float tile[32][33];
  const size_t base = (size_t)blockIdx.z * (size_t)R * (size_t)C;
  const int c0 = blockIdx.x * 32, r0 = blockIdx.y * 32;
  for (int j = threadIdx.y; j < 32; j += 8)
    tile[j][threadIdx.x] = in[base + (size_t)(r0 + j) * C + c0 + threadIdx.x];
  __syncthreads();
  for (int j = threadIdx.y; j < 32; j += 8)
    out[base + (size_t)(c0 + j) * R + r0 + threadIdx.x] =
        (bf16_t)tile[threadIdx.x][j];
}

// ---------------------------------------------------------------------------
// LayerNorm (weight only, eps=1e-5) over last dim 256, output bf16.
// One wave per row, 8 rows per block.
// ---------------------------------------------------------------------------
__global__ __launch_bounds__(256)
void ln_kernel(const float* __restrict__ x, const float* __restrict__ w,
               bf16_t* __restrict__ out) {
  const int lane = threadIdx.x & 31;
  const int wave = threadIdx.x >> 5;
  const size_t row = (size_t)blockIdx.x * 8 + wave;
  const float* xr = x + row * 256 + lane * 8;
  v4f a = *(const v4f*)xr;
  v4f b = *(const v4f*)(xr + 4);
  float s = 0.f, s2 = 0.f;
#pragma unroll
  for (int i = 0; i < 4; ++i) {
    s  += a[i] + b[i];
    s2 += a[i] * a[i] + b[i] * b[i];
  }
#pragma unroll
  for (int off = 16; off > 0; off >>= 1) {
    s  += __shfl_xor(s,  off, 32);
    s2 += __shfl_xor(s2, off, 32);
  }
  const float mu  = s * (1.0f / 256.0f);
  const float var = s2 * (1.0f / 256.0f) - mu * mu;
  const float r   = rsqrtf(var + 1e-5f);
  const float* wr = w + lane * 8;
  v8bf o;
#pragma unroll
  for (int i = 0; i < 4; ++i) {
    o[i]     = (bf16_t)((a[i] - mu) * r * wr[i]);
    o[4 + i] = (bf16_t)((b[i] - mu) * r * wr[4 + i]);
  }
  *(v8bf*)(out + row * 256 + lane * 8) = o;
}

// ---------------------------------------------------------------------------
// Generic WMMA GEMM:  C[M,N] = epi( A[M,K] @ BT[N,K]^T )
//   AMODE 0: A is bf16        AMODE 1: A is fp32, converted to bf16 on load
//   EPI   0: bf16 out, +bias, exact GELU
//   EPI   1: bf16 out, +bias
//   EPI   2: fp32 out, *scale
// Block tile 128x256, k-step 32; 8 waves in 2x4, wave tile 64x64 (16 WMMA
// accumulators). Double-buffered LDS; async global->LDS DMA when available.
// ---------------------------------------------------------------------------
template <int AMODE, int EPI>
__global__ __launch_bounds__(256)
void gemm_kernel(const void* __restrict__ Aptr, const bf16_t* __restrict__ BT,
                 const float* __restrict__ bias, void* __restrict__ Cptr,
                 int M, int N, int K,
                 long long sA, long long sB, long long sC, float scale) {
  constexpr int BM = 128, BN = 256;
  __shared__ bf16_t lsA[2][BM * LDL];
  __shared__ bf16_t lsB[2][BN * LDL];

  const int tid  = threadIdx.x;
  const int lane = tid & 31;
  const int wave = tid >> 5;
  const int wm = wave & 1;   // 0..1 -> 64 rows
  const int wn = wave >> 1;  // 0..3 -> 64 cols
  const int lh = lane >> 4;  // 0/1
  const int lm = lane & 15;

  const int bm = blockIdx.y * BM;
  const int bn = blockIdx.x * BN;

  const bf16_t* Ab = (const bf16_t*)Aptr + (size_t)blockIdx.z * sA;
  const float*  Af = (const float*)Aptr  + (size_t)blockIdx.z * sA;
  const bf16_t* Bb = BT + (size_t)blockIdx.z * sB;

  // copy mapping: b128 chunks (8 bf16); 4 chunks per 32-wide row
  const int crow = tid >> 2;        // 0..63
  const int ccol = (tid & 3) * 8;   // 0,8,16,24

  v8bf ra[2];   // staging (fallback and fp32-A path)
  v8bf rb[4];

  auto issueA = [&](int kt, int buf) {
#pragma unroll
    for (int j = 0; j < 2; ++j) {
      const int row = crow + 64 * j;
      const size_t off = (size_t)(bm + row) * K + kt + ccol;
      if (AMODE == 0) {
#if HAVE_ASYNC_LDS
        __builtin_amdgcn_global_load_async_to_lds_b128(
            (as1_v4i*)(Ab + off), (as3_v4i*)(&lsA[buf][row * LDL + ccol]), 0, 0);
#else
        ra[j] = *(const v8bf*)(Ab + off);
#endif
      } else {
        v4f f0 = *(const v4f*)(Af + off);
        v4f f1 = *(const v4f*)(Af + off + 4);
#pragma unroll
        for (int i = 0; i < 4; ++i) {
          ra[j][i]     = (bf16_t)f0[i];
          ra[j][4 + i] = (bf16_t)f1[i];
        }
      }
    }
  };
  auto issueB = [&](int kt, int buf) {
#pragma unroll
    for (int j = 0; j < 4; ++j) {
      const int row = crow + 64 * j;
      const size_t off = (size_t)(bn + row) * K + kt + ccol;
#if HAVE_ASYNC_LDS
      __builtin_amdgcn_global_load_async_to_lds_b128(
          (as1_v4i*)(Bb + off), (as3_v4i*)(&lsB[buf][row * LDL + ccol]), 0, 0);
#else
      rb[j] = *(const v8bf*)(Bb + off);
#endif
    }
  };
  auto commit = [&](int buf) {  // register -> LDS for data not copied async
    if (AMODE == 1 || !HAVE_ASYNC_LDS) {
#pragma unroll
      for (int j = 0; j < 2; ++j)
        *(v8bf*)(&lsA[buf][(crow + 64 * j) * LDL + ccol]) = ra[j];
    }
#if !HAVE_ASYNC_LDS
#pragma unroll
    for (int j = 0; j < 4; ++j)
      *(v8bf*)(&lsB[buf][(crow + 64 * j) * LDL + ccol]) = rb[j];
#endif
  };

  v8f acc[4][4];
  {
    v8f z = {0.f, 0.f, 0.f, 0.f, 0.f, 0.f, 0.f, 0.f};
#pragma unroll
    for (int mi = 0; mi < 4; ++mi)
#pragma unroll
      for (int ni = 0; ni < 4; ++ni) acc[mi][ni] = z;
  }

  issueA(0, 0);
  issueB(0, 0);
  commit(0);
#if HAVE_ASYNC_LDS
  __builtin_amdgcn_s_wait_asynccnt(0);
#endif
  __syncthreads();

  const int nk = K / 32;
  for (int ik = 0; ik < nk; ++ik) {
    const int cur = ik & 1;
    if (ik + 1 < nk) {             // next tile's copies in flight
      issueA((ik + 1) * 32, cur ^ 1);
      issueB((ik + 1) * 32, cur ^ 1);
    }

    const bf16_t* As = &lsA[cur][(wm * 64) * LDL];
    const bf16_t* Bs = &lsB[cur][(wn * 64) * LDL];

    // B fragments (dense 32x16): lane n -> column n; lane half -> K 0-15/16-31
    v16bf bfr[4];
#pragma unroll
    for (int ni = 0; ni < 4; ++ni) {
      const bf16_t* p = Bs + (ni * 16 + lm) * LDL + lh * 16;
      union { v16bf v; v8bf h[2]; } u;
      u.h[0] = *(const v8bf*)(p);
      u.h[1] = *(const v8bf*)(p + 8);
      bfr[ni] = u.v;
    }
    // A fragments (16x32): lane m -> row m; lane half -> K 0-15/16-31
#pragma unroll
    for (int mi = 0; mi < 4; ++mi) {
      const bf16_t* p = As + (mi * 16 + lm) * LDL + lh * 16;
      union { v16bf v; v8bf h[2]; } u;
      u.h[0] = *(const v8bf*)(p);
      u.h[1] = *(const v8bf*)(p + 8);
      v16bf afr = u.v;
#pragma unroll
      for (int ni = 0; ni < 4; ++ni)
        acc[mi][ni] = __builtin_amdgcn_wmma_f32_16x16x32_bf16(
            false, afr, false, bfr[ni], (short)0, acc[mi][ni], false, false);
    }

    if (ik + 1 < nk) commit(cur ^ 1);
#if HAVE_ASYNC_LDS
    __builtin_amdgcn_s_wait_asynccnt(0);
#endif
    __syncthreads();
  }

  // Epilogue. C frag layout: lane -> N = lane%16, VGPR v -> M = v + (lane/16)*8
  const int row0 = bm + wm * 64 + lh * 8;
  const int col0 = bn + wn * 64;
#pragma unroll
  for (int ni = 0; ni < 4; ++ni) {
    const int n = col0 + ni * 16 + lm;
    float bi = 0.0f;
    if (EPI < 2) bi = bias[n];
#pragma unroll
    for (int mi = 0; mi < 4; ++mi) {
      const int mrow = row0 + mi * 16;
#pragma unroll
      for (int v = 0; v < 8; ++v) {
        float xv = acc[mi][ni][v] + bi;
        if (EPI == 0)  // exact GELU: 0.5x(1+erf(x/sqrt(2)))
          xv = 0.5f * xv * (1.0f + erff(xv * 0.70710678118654752f));
        const size_t idx =
            (size_t)blockIdx.z * sC + (size_t)(mrow + v) * N + n;
        if (EPI < 2) ((bf16_t*)Cptr)[idx] = (bf16_t)xv;
        else         ((float*)Cptr)[idx]  = xv * scale;
      }
    }
  }
}

// ---------------------------------------------------------------------------
// Row softmax in place, 2048 cols, one 256-thread block per row.
// ---------------------------------------------------------------------------
__global__ __launch_bounds__(256)
void softmax_kernel(float* __restrict__ attn) {
  const int lane = threadIdx.x & 31;
  const int wave = threadIdx.x >> 5;
  float* p = attn + (size_t)blockIdx.x * 2048 + threadIdx.x * 8;
  v4f a = *(const v4f*)p;
  v4f b = *(const v4f*)(p + 4);

  __shared__ float red[8];
  float m = -3.402823466e38f;
#pragma unroll
  for (int i = 0; i < 4; ++i) m = fmaxf(m, fmaxf(a[i], b[i]));
#pragma unroll
  for (int off = 16; off > 0; off >>= 1) m = fmaxf(m, __shfl_xor(m, off, 32));
  if (lane == 0) red[wave] = m;
  __syncthreads();
  float bm = red[0];
#pragma unroll
  for (int i = 1; i < 8; ++i) bm = fmaxf(bm, red[i]);

  float s = 0.f;
#pragma unroll
  for (int i = 0; i < 4; ++i) {
    a[i] = __expf(a[i] - bm); s += a[i];
    b[i] = __expf(b[i] - bm); s += b[i];
  }
#pragma unroll
  for (int off = 16; off > 0; off >>= 1) s += __shfl_xor(s, off, 32);
  __syncthreads();
  if (lane == 0) red[wave] = s;
  __syncthreads();
  float bs = 0.f;
#pragma unroll
  for (int i = 0; i < 8; ++i) bs += red[i];
  const float inv = 1.0f / bs;
#pragma unroll
  for (int i = 0; i < 4; ++i) { a[i] *= inv; b[i] *= inv; }
  *(v4f*)p = a;
  *(v4f*)(p + 4) = b;
}

// ---------------------------------------------------------------------------
// Host launcher
// ---------------------------------------------------------------------------
extern "C" void kernel_launch(void* const* d_in, const int* in_sizes, int n_in,
                              void* d_out, int out_size, void* d_ws,
                              size_t ws_size, hipStream_t stream) {
  (void)in_sizes; (void)n_in; (void)out_size; (void)ws_size;
  constexpr int B = 32, Lq = 512, Lk = 2048, Cin = 256, HID = 1024, E = 512, Dv = 256;
  constexpr int Tq = B * Lq;   // 16384
  constexpr int Tk = B * Lk;   // 65536

  const float* q_x      = (const float*)d_in[0];
  const float* k_x      = (const float*)d_in[1];
  const float* v_x      = (const float*)d_in[2];
  const float* ln_q_w   = (const float*)d_in[3];
  const float* ln_k_w   = (const float*)d_in[4];
  const float* q_fc_w   = (const float*)d_in[5];
  const float* q_fc_b   = (const float*)d_in[6];
  const float* q_proj_w = (const float*)d_in[7];
  const float* q_proj_b = (const float*)d_in[8];
  const float* k_fc_w   = (const float*)d_in[9];
  const float* k_fc_b   = (const float*)d_in[10];
  const float* k_proj_w = (const float*)d_in[11];
  const float* k_proj_b = (const float*)d_in[12];

  float* x_trend = (float*)d_out;                        // [B,Lq,Dv]
  float* attn    = x_trend + (size_t)B * Lq * Dv;        // [B,Lq,Lk]

  // workspace carve-up (bf16)
  bf16_t* w = (bf16_t*)d_ws;
  bf16_t* qfcT   = w; w += (size_t)HID * Cin;
  bf16_t* qprojT = w; w += (size_t)E * HID;
  bf16_t* kfcT   = w; w += (size_t)HID * Cin;
  bf16_t* kprojT = w; w += (size_t)E * HID;
  bf16_t* vT     = w; w += (size_t)B * Dv * Lk;
  bf16_t* xln    = w; w += (size_t)Tk * Cin;
  bf16_t* h      = w; w += (size_t)Tk * HID;
  bf16_t* qe     = w; w += (size_t)Tq * E;
  bf16_t* ke     = w; w += (size_t)Tk * E;

  const dim3 tb(32, 8);
  transpose_cast<<<dim3(HID / 32, Cin / 32, 1), tb, 0, stream>>>(q_fc_w, qfcT, Cin, HID);
  transpose_cast<<<dim3(E / 32, HID / 32, 1),  tb, 0, stream>>>(q_proj_w, qprojT, HID, E);
  transpose_cast<<<dim3(HID / 32, Cin / 32, 1), tb, 0, stream>>>(k_fc_w, kfcT, Cin, HID);
  transpose_cast<<<dim3(E / 32, HID / 32, 1),  tb, 0, stream>>>(k_proj_w, kprojT, HID, E);
  transpose_cast<<<dim3(Dv / 32, Lk / 32, B),  tb, 0, stream>>>(v_x, vT, Lk, Dv);

  // ---- Q path ----
  ln_kernel<<<Tq / 8, 256, 0, stream>>>(q_x, ln_q_w, xln);
  gemm_kernel<0, 0><<<dim3(HID / 256, Tq / 128, 1), 256, 0, stream>>>(
      xln, qfcT, q_fc_b, h, Tq, HID, Cin, 0, 0, 0, 1.0f);
  gemm_kernel<0, 1><<<dim3(E / 256, Tq / 128, 1), 256, 0, stream>>>(
      h, qprojT, q_proj_b, qe, Tq, E, HID, 0, 0, 0, 1.0f);

  // ---- K path ----
  ln_kernel<<<Tk / 8, 256, 0, stream>>>(k_x, ln_k_w, xln);
  gemm_kernel<0, 0><<<dim3(HID / 256, Tk / 128, 1), 256, 0, stream>>>(
      xln, kfcT, k_fc_b, h, Tk, HID, Cin, 0, 0, 0, 1.0f);
  gemm_kernel<0, 1><<<dim3(E / 256, Tk / 128, 1), 256, 0, stream>>>(
      h, kprojT, k_proj_b, ke, Tk, E, HID, 0, 0, 0, 1.0f);

  // ---- scores = Q K^T * (1/sqrt(E)), fp32 straight into d_out ----
  gemm_kernel<0, 2><<<dim3(Lk / 256, Lq / 128, B), 256, 0, stream>>>(
      qe, ke, nullptr, attn, Lq, Lk, E,
      (long long)Lq * E, (long long)Lk * E, (long long)Lq * Lk,
      0.04419417382415922f /* 1/sqrt(512) */);

  // ---- softmax rows in place ----
  softmax_kernel<<<B * Lq, 256, 0, stream>>>(attn);

  // ---- x_trend = attn @ V  (A fp32 -> bf16 on load, B = V^T bf16) ----
  gemm_kernel<1, 2><<<dim3(Dv / 256, Lq / 128, B), 256, 0, stream>>>(
      attn, vT, nullptr, x_trend, Lq, Dv, Lk,
      (long long)Lq * Lk, (long long)Dv * Lk, (long long)Lq * Dv, 1.0f);
}
